// CausalAttNet_45449343926922
// MI455X (gfx1250) — matrix operand order repaired and checked by reading, not compiled, not run on hardware
//
#include <hip/hip_runtime.h>
#include <hip/hip_bf16.h>
#include <stddef.h>

// Problem constants (from reference)
#define NNODES 65536
#define HIDDIM 256
#define NEDGES 524288
#define NGRAPH 64
#define EPERG  8192
#define NRESK  4096
#define ETOT   262144  // NGRAPH * NRESK

typedef __attribute__((ext_vector_type(16))) __bf16 v16bf;
typedef __attribute__((ext_vector_type(8)))  float  v8f;
typedef __attribute__((ext_vector_type(4)))  int    v4i;

union AFrag { v16bf v; __bf16 e[16]; };

// Output layout (flat floats, reference tuple order)
static constexpr size_t O_CX  = 0;
static constexpr size_t O_CEI = O_CX  + (size_t)NNODES * HIDDIM;
static constexpr size_t O_CW  = O_CEI + 2 * (size_t)ETOT;
static constexpr size_t O_CB  = O_CW  + (size_t)ETOT;
static constexpr size_t O_FX  = O_CB  + (size_t)NNODES;
static constexpr size_t O_FEI = O_FX  + (size_t)NNODES * HIDDIM;
static constexpr size_t O_FW  = O_FEI + 2 * (size_t)ETOT;
static constexpr size_t O_FB  = O_FW  + (size_t)ETOT;
static constexpr size_t O_ES  = O_FB  + (size_t)NNODES;

#if defined(__gfx1250__) && \
    __has_builtin(__builtin_amdgcn_global_load_async_to_lds_b128) && \
    __has_builtin(__builtin_amdgcn_s_wait_asynccnt)
#define USE_ASYNC_LDS 1
#else
#define USE_ASYNC_LDS 0
#endif

// ---------------------------------------------------------------- utilities
__global__ void zero_u32(unsigned* __restrict__ p, size_t n) {
  size_t t = (size_t)blockIdx.x * 256 + threadIdx.x;
  size_t stride = (size_t)gridDim.x * 256;
  for (; t < n; t += stride) p[t] = 0u;
}

// Pack fp32 W (K=256 x N=256, row-major) into bf16 WMMA B-fragment order:
// fragment f = kt*16 + ntile; per fragment 32 lanes x 16 bf16 contiguous.
// Lane l<16: col = ntile*16+l, K = kt*32 + j (j=0..15)
// Lane l>=16: col = ntile*16+(l-16), K = kt*32 + 16 + j
__global__ void pack_w(const float* __restrict__ W, __bf16* __restrict__ Wp) {
  int t = blockIdx.x * blockDim.x + threadIdx.x;   // 0..4095 (128 frags * 32 lanes)
  if (t >= 128 * 32) return;
  int lane  = t & 31;
  int f     = t >> 5;
  int kt    = f >> 4;
  int ntile = f & 15;
  int colB  = ntile * 16 + (lane & 15);
  int kBase = kt * 32 + (lane >> 4) * 16;
  __bf16* dst = Wp + (size_t)t * 16;
  for (int j = 0; j < 16; ++j)
    dst[j] = (__bf16)W[(size_t)(kBase + j) * HIDDIM + colB];
}

// ---------------------------------------------------------------- scatter add
__global__ __launch_bounds__(256)
void scatter_add(const float* __restrict__ H, const int* __restrict__ row,
                 const int* __restrict__ col, float* __restrict__ Agg) {
  const int e = blockIdx.x;
  const int c = threadIdx.x;
  const int r = row[e], s = col[e];
  atomicAdd(&Agg[(size_t)r * HIDDIM + c], H[(size_t)s * HIDDIM + c]);
}

// ---------------------------------------------------------------- WMMA GEMM
// Y[64rows x 256] = relu( (X (+Agg)) @ Wpacked + bias ), bf16 inputs, f32 acc.
__global__ __launch_bounds__(256)
void gin_gemm_relu(const float* __restrict__ X, const float* __restrict__ Agg,
                   const __bf16* __restrict__ Wp, const float* __restrict__ bias,
                   float* __restrict__ Y) {
  __shared__ float sA[64 * HIDDIM];  // 64 KB
  const int tid  = threadIdx.x;
  const int wave = tid >> 5;
  const int lane = tid & 31;
  const size_t rowBase = (size_t)blockIdx.x * 64;

  if (tid == 0)  // warm next block's A tile into L2 (global_prefetch_b8)
    __builtin_prefetch(X + (rowBase + 64) * HIDDIM, 0, 1);

  // Stage A tile in LDS (fp32). No-Agg case: pure byte copy -> use the
  // CDNA5 async global->LDS datapath (ASYNCcnt, no VGPR round trip).
  if (Agg == nullptr) {
#if USE_ASYNC_LDS
    const v4i* gsrc = (const v4i*)(X + rowBase * HIDDIM);
    __attribute__((address_space(3))) v4i* ldst =
        (__attribute__((address_space(3))) v4i*)sA;
    for (int i = tid; i < 64 * (HIDDIM / 4); i += 256)
      __builtin_amdgcn_global_load_async_to_lds_b128(
          (__attribute__((address_space(1))) v4i*)(gsrc + i),
          ldst + i, 0, 0);
    __builtin_amdgcn_s_wait_asynccnt(0);
#else
    const float4* Xv = (const float4*)(X + rowBase * HIDDIM);
    float4* Sv = (float4*)sA;
    for (int i = tid; i < 64 * (HIDDIM / 4); i += 256) Sv[i] = Xv[i];
#endif
  } else {
    const float4* Xv = (const float4*)(X + rowBase * HIDDIM);
    const float4* Av = (const float4*)(Agg + rowBase * HIDDIM);
    float4* Sv = (float4*)sA;
    for (int i = tid; i < 64 * (HIDDIM / 4); i += 256) {
      float4 v = Xv[i];
      float4 a = Av[i];
      v.x += a.x; v.y += a.y; v.z += a.z; v.w += a.w;
      Sv[i] = v;
    }
  }
  __syncthreads();

  const int rtile = wave & 3;   // 4 row-tiles of 16
  const int chalf = wave >> 2;  // 2 column halves of 128
  const int m     = lane & 15;
  const int khalf = lane >> 4;
  const float* aRow = &sA[(rtile * 16 + m) * HIDDIM];

  v8f zero = {0.f, 0.f, 0.f, 0.f, 0.f, 0.f, 0.f, 0.f};
  v8f acc[8];
#pragma unroll
  for (int t = 0; t < 8; ++t) acc[t] = zero;

  for (int kt = 0; kt < 8; ++kt) {
    const int k0 = kt * 32;
    AFrag a;
#pragma unroll
    for (int j = 0; j < 16; ++j) a.e[j] = (__bf16)aRow[k0 + khalf * 16 + j];

    const __bf16* wk = Wp + ((size_t)(kt * 16 + chalf * 8) * 512) + (size_t)lane * 16;
#pragma unroll
    for (int nt = 0; nt < 8; ++nt) {
      v16bf bfr = *(const v16bf*)(wk + (size_t)nt * 512);
      acc[nt] = __builtin_amdgcn_wmma_f32_16x16x32_bf16(
          false, a.v, false, bfr, (short)0, acc[nt], false, false);
    }
  }

  // Epilogue: C/D layout — VGPR r: lanes 0-15 -> M=r, lanes 16-31 -> M=8+r
#pragma unroll
  for (int nt = 0; nt < 8; ++nt) {
    const int colIdx = chalf * 128 + nt * 16 + m;
    const float bv = bias[colIdx];
#pragma unroll
    for (int r = 0; r < 8; ++r) {
      const int mrow = rtile * 16 + khalf * 8 + r;
      float v = acc[nt][r] + bv;
      v = v > 0.f ? v : 0.f;
      Y[(rowBase + mrow) * HIDDIM + colIdx] = v;
    }
  }
}

// ---------------------------------------------------------------- scoring
__global__ __launch_bounds__(256)
void score_rows(const float* __restrict__ H, const float* __restrict__ wsc,
                float* __restrict__ ssrc, float* __restrict__ sdst) {
  const int wave = threadIdx.x >> 5, lane = threadIdx.x & 31;
  const size_t r = (size_t)blockIdx.x * 8 + wave;
  const float* hp = H + r * HIDDIM;
  float a = 0.f, b = 0.f;
  for (int j = lane; j < HIDDIM; j += 32) {
    float h = hp[j];
    a += h * wsc[j];
    b += h * wsc[HIDDIM + j];
  }
  for (int o = 16; o; o >>= 1) { a += __shfl_down(a, o, 32); b += __shfl_down(b, o, 32); }
  if (lane == 0) { ssrc[r] = a; sdst[r] = b; }
}

__global__ void edge_scores(const int* __restrict__ row, const int* __restrict__ col,
                            const float* __restrict__ ssrc, const float* __restrict__ sdst,
                            const float* __restrict__ bsc,
                            float* __restrict__ esWs, float* __restrict__ outEs) {
  size_t t = (size_t)blockIdx.x * 256 + threadIdx.x;
  if (t >= NEDGES) return;
  float v = ssrc[row[t]] + sdst[col[t]] + bsc[0];
  esWs[t] = v;
  outEs[t] = v;
}

// ---------------------------------------------------------------- per-graph sort
__global__ __launch_bounds__(1024)
void sort_graph(const float* __restrict__ es, int* __restrict__ order) {
  __shared__ float sk[EPERG];  // 32 KB
  __shared__ int   si[EPERG];  // 32 KB
  const int b = blockIdx.x;
  for (int i = threadIdx.x; i < EPERG; i += 1024) { sk[i] = es[(size_t)b * EPERG + i]; si[i] = i; }
  __syncthreads();
  for (int size = 2; size <= EPERG; size <<= 1) {
    for (int stride = size >> 1; stride > 0; stride >>= 1) {
      for (int i = threadIdx.x; i < EPERG / 2; i += 1024) {
        int lo = 2 * i - (i & (stride - 1));
        int hi = lo + stride;
        bool desc = ((lo & size) == 0);
        float kl = sk[lo], kh = sk[hi];
        int il = si[lo], ih = si[hi];
        bool bad = desc ? (kl < kh || (kl == kh && il > ih))
                        : (kl > kh || (kl == kh && il < ih));
        if (bad) { sk[lo] = kh; sk[hi] = kl; si[lo] = ih; si[hi] = il; }
      }
      __syncthreads();
    }
  }
  for (int i = threadIdx.x; i < EPERG; i += 1024) order[(size_t)b * EPERG + i] = si[i];
}

__global__ void select_edges(const int* __restrict__ ei0, const int* __restrict__ ei1,
                             const int* __restrict__ order, const float* __restrict__ es,
                             float* __restrict__ out,
                             int* __restrict__ cE0, int* __restrict__ cE1,
                             int* __restrict__ fE0, int* __restrict__ fE1) {
  size_t t = (size_t)blockIdx.x * 256 + threadIdx.x;
  if (t >= NEDGES) return;
  int b = (int)(t / EPERG), j = (int)(t % EPERG);
  size_t src = (size_t)b * EPERG + order[t];
  int e0 = ei0[src], e1 = ei1[src];
  float sc = es[src];
  if (j < NRESK) {
    size_t p = (size_t)b * NRESK + j;
    out[O_CW + p] = sc;
    cE0[p] = e0; cE1[p] = e1;
  } else {
    size_t p = (size_t)b * NRESK + (j - NRESK);
    out[O_FW + p] = -sc;
    fE0[p] = e0; fE1[p] = e1;
  }
}

// ---------------------------------------------------------------- relabel
__global__ void mark_presence(const int* __restrict__ E0, const int* __restrict__ E1,
                              int* __restrict__ pres) {
  size_t t = (size_t)blockIdx.x * 256 + threadIdx.x;
  if (t >= ETOT) return;
  pres[E0[t]] = 1;
  pres[E1[t]] = 1;
}

__global__ __launch_bounds__(1024)
void scan_compact(const int* __restrict__ pres, int* __restrict__ nodeIdx,
                  int* __restrict__ sub, int* __restrict__ cnt) {
  __shared__ int ssum[1024];
  const int t = threadIdx.x;
  const int base = t * 64;
  int s = 0;
  for (int i = 0; i < 64; ++i) s += pres[base + i];
  ssum[t] = s;
  __syncthreads();
  for (int o = 1; o < 1024; o <<= 1) {
    int v = (t >= o) ? ssum[t - o] : 0;
    __syncthreads();
    ssum[t] += v;
    __syncthreads();
  }
  int pos = ssum[t] - s;  // exclusive prefix for this chunk
  for (int i = 0; i < 64; ++i) {
    int n = base + i;
    if (pres[n]) { nodeIdx[n] = pos; sub[pos] = n; ++pos; }
  }
  if (t == 1023) cnt[0] = ssum[1023];
}

__global__ void emit_ei_rel(const int* __restrict__ E0, const int* __restrict__ E1,
                            const int* __restrict__ nodeIdx, float* __restrict__ outBase) {
  size_t t = (size_t)blockIdx.x * 256 + threadIdx.x;
  if (t >= ETOT) return;
  outBase[t]        = (float)nodeIdx[E0[t]];
  outBase[ETOT + t] = (float)nodeIdx[E1[t]];
}

__global__ __launch_bounds__(256)
void emit_x_batch(const float* __restrict__ H, const int* __restrict__ sub,
                  const int* __restrict__ cnt, const int* __restrict__ batch,
                  float* __restrict__ outX, float* __restrict__ outB) {
  const int rowp = blockIdx.x, c = threadIdx.x;
  const int n = cnt[0];
  float v = 0.f;
  int srow = 0;
  if (rowp < n) { srow = sub[rowp]; v = H[(size_t)srow * HIDDIM + c]; }
  outX[(size_t)rowp * HIDDIM + c] = v;
  if (c == 0) outB[rowp] = (rowp < n) ? (float)batch[srow] : -1.0f;
}

// ---------------------------------------------------------------- launch
extern "C" void kernel_launch(void* const* d_in, const int* in_sizes, int n_in,
                              void* d_out, int out_size, void* d_ws, size_t ws_size,
                              hipStream_t stream) {
  (void)in_sizes; (void)n_in; (void)out_size; (void)ws_size;

  const float* x    = (const float*)d_in[0];
  const int*   ei   = (const int*)d_in[1];
  const int*   row  = ei;
  const int*   col  = ei + NEDGES;
  const int*   bat  = (const int*)d_in[2];
  const float* W11  = (const float*)d_in[3];  const float* b11 = (const float*)d_in[4];
  const float* W12  = (const float*)d_in[5];  const float* b12 = (const float*)d_in[6];
  const float* W21  = (const float*)d_in[7];  const float* b21 = (const float*)d_in[8];
  const float* W22  = (const float*)d_in[9];  const float* b22 = (const float*)d_in[10];
  const float* wsc  = (const float*)d_in[11];
  const float* bsc  = (const float*)d_in[12];
  float* out = (float*)d_out;

  char* wsb = (char*)d_ws;
  size_t cur = 0;
  auto take = [&](size_t bytes) -> void* {
    void* p = wsb + cur;
    cur = (cur + bytes + 255) & ~(size_t)255;
    return p;
  };
  const size_t HB = (size_t)NNODES * HIDDIM * sizeof(float);  // 64 MB
  __bf16* wp1 = (__bf16*)take(131072);
  __bf16* wp2 = (__bf16*)take(131072);
  __bf16* wp3 = (__bf16*)take(131072);
  __bf16* wp4 = (__bf16*)take(131072);
  float* agg  = (float*)take(HB);
  float* bufA = (float*)take(HB);
  float* bufB = (float*)take(HB);
  float* ssrc = (float*)take((size_t)NNODES * 4);
  float* sdst = (float*)take((size_t)NNODES * 4);
  float* esWs = (float*)take((size_t)NEDGES * 4);
  int* order  = (int*)take((size_t)NEDGES * 4);
  int* cE0 = (int*)take((size_t)ETOT * 4);
  int* cE1 = (int*)take((size_t)ETOT * 4);
  int* fE0 = (int*)take((size_t)ETOT * 4);
  int* fE1 = (int*)take((size_t)ETOT * 4);
  int* presC = (int*)take((size_t)NNODES * 4);
  int* nidxC = (int*)take((size_t)NNODES * 4);
  int* subC  = (int*)take((size_t)NNODES * 4);
  int* cntC  = (int*)take(256);
  int* presF = (int*)take((size_t)NNODES * 4);
  int* nidxF = (int*)take((size_t)NNODES * 4);
  int* subF  = (int*)take((size_t)NNODES * 4);
  int* cntF  = (int*)take(256);

  // Pack all weights to bf16 fragment layout
  pack_w<<<16, 256, 0, stream>>>(W11, wp1);
  pack_w<<<16, 256, 0, stream>>>(W12, wp2);
  pack_w<<<16, 256, 0, stream>>>(W21, wp3);
  pack_w<<<16, 256, 0, stream>>>(W22, wp4);

  // GIN layer 1
  zero_u32<<<4096, 256, 0, stream>>>((unsigned*)agg, (size_t)NNODES * HIDDIM);
  scatter_add<<<NEDGES, 256, 0, stream>>>(x, row, col, agg);
  gin_gemm_relu<<<NNODES / 64, 256, 0, stream>>>(x, agg, wp1, b11, bufA);
  gin_gemm_relu<<<NNODES / 64, 256, 0, stream>>>(bufA, nullptr, wp2, b12, bufB);

  // GIN layer 2 (final h in bufB)
  zero_u32<<<4096, 256, 0, stream>>>((unsigned*)agg, (size_t)NNODES * HIDDIM);
  scatter_add<<<NEDGES, 256, 0, stream>>>(bufB, row, col, agg);
  gin_gemm_relu<<<NNODES / 64, 256, 0, stream>>>(bufB, agg, wp3, b21, bufA);
  gin_gemm_relu<<<NNODES / 64, 256, 0, stream>>>(bufA, nullptr, wp4, b22, bufB);

  // Scores
  score_rows<<<NNODES / 8, 256, 0, stream>>>(bufB, wsc, ssrc, sdst);
  edge_scores<<<NEDGES / 256, 256, 0, stream>>>(row, col, ssrc, sdst, bsc, esWs, out + O_ES);

  // Per-graph descending argsort + split
  sort_graph<<<NGRAPH, 1024, 0, stream>>>(esWs, order);
  select_edges<<<NEDGES / 256, 256, 0, stream>>>(row, col, order, esWs, out, cE0, cE1, fE0, fE1);

  // Relabel (causal)
  zero_u32<<<256, 256, 0, stream>>>((unsigned*)presC, NNODES);
  mark_presence<<<ETOT / 256, 256, 0, stream>>>(cE0, cE1, presC);
  scan_compact<<<1, 1024, 0, stream>>>(presC, nidxC, subC, cntC);
  emit_ei_rel<<<ETOT / 256, 256, 0, stream>>>(cE0, cE1, nidxC, out + O_CEI);
  emit_x_batch<<<NNODES, 256, 0, stream>>>(bufB, subC, cntC, bat, out + O_CX, out + O_CB);

  // Relabel (conf)
  zero_u32<<<256, 256, 0, stream>>>((unsigned*)presF, NNODES);
  mark_presence<<<ETOT / 256, 256, 0, stream>>>(fE0, fE1, presF);
  scan_compact<<<1, 1024, 0, stream>>>(presF, nidxF, subF, cntF);
  emit_ei_rel<<<ETOT / 256, 256, 0, stream>>>(fE0, fE1, nidxF, out + O_FEI);
  emit_x_batch<<<NNODES, 256, 0, stream>>>(bufB, subF, cntF, bat, out + O_FX, out + O_FB);
}
